// CausalSelfAttention_86844238725187
// MI455X (gfx1250) — matrix operand order repaired and checked by reading, not compiled, not run on hardware
//
#include <hip/hip_runtime.h>
#include <hip/hip_bf16.h>
#include <math.h>

#define T_SEQ   4096
#define DM      1024
#define NHEAD   16
#define DH      64

typedef __bf16 bf16;
typedef __attribute__((ext_vector_type(16))) __bf16 v16bf;
typedef __attribute__((ext_vector_type(8)))  __bf16 v8bf;
typedef __attribute__((ext_vector_type(4)))  __bf16 v4bf;
typedef __attribute__((ext_vector_type(8)))  float  v8f;
typedef __attribute__((ext_vector_type(4)))  float  v4f;
typedef __attribute__((ext_vector_type(4)))  unsigned int u32x4;
typedef __attribute__((ext_vector_type(8)))  int    i32x8;
typedef __attribute__((ext_vector_type(4)))  int    i32x4;

static __device__ __forceinline__ void wave_lds_fence() {
  asm volatile("s_wait_dscnt 0" ::: "memory");
}

static __device__ __forceinline__ v8f wmma_bf16(v16bf a, v16bf b, v8f c) {
  return __builtin_amdgcn_wmma_f32_16x16x32_bf16(false, a, false, b, (short)0, c,
                                                 false, false);
}

static __device__ __forceinline__ v8f zero8() {
  v8f z;
#pragma unroll
  for (int e = 0; e < 8; ++e) z[e] = 0.f;
  return z;
}

static __device__ __forceinline__ v16bf cat16(v8bf lo, v8bf hi) {
  return __builtin_shufflevector(lo, hi, 0, 1, 2, 3, 4, 5, 6, 7,
                                 8, 9, 10, 11, 12, 13, 14, 15);
}

// ---- Fragment loaders (ISA 7.12.2, wave32), A via 16B vector loads, B via scalar loads ------
// A 16x32 bf16: lane -> row m=l&15 ; elem e -> k = (l>=16?8:0)+e+(e>=8?8:0)
static __device__ __forceinline__ v16bf load_a16(const bf16* p, int ld, int lane) {
  const bf16* row = p + (lane & 15) * ld + ((lane >> 4) << 3);
  v8bf lo = *(const v8bf*)(row);        // k = base+0..7
  v8bf hi = *(const v8bf*)(row + 16);   // k = base+16..23
  return cat16(lo, hi);
}
// B 32x16 k-major: lane -> n=l&15 ; elem e -> k = (l>=16?8:0)+e+(e>=8?8:0)
static __device__ __forceinline__ v16bf load_b16(const bf16* p, int ld, int lane) {
  const bf16* col = p + (lane & 15) + ((lane >> 4) << 3) * ld;
  v16bf b;
#pragma unroll
  for (int s = 0; s < 8; ++s) {
    b[s]     = col[s * ld];
    b[s + 8] = col[(16 + s) * ld];
  }
  return b;
}

// ---- TDM: 2D bf16 tile (rows x rowlen) global -> LDS, packed --------------
// D# per ISA 8.3/8.4. 6-arg builtin on this toolchain.
static __device__ __forceinline__ void tdm_load_2d_bf16(unsigned lds_off,
                                                        const void* gptr,
                                                        unsigned rows,
                                                        unsigned rowlen,
                                                        unsigned stride) {
  unsigned long long ga = (unsigned long long)gptr;
  u32x4 g0;
  g0[0] = 1u;                                            // count=1 (user D#)
  g0[1] = lds_off;                                       // lds_addr
  g0[2] = (unsigned)ga;                                  // global_addr[31:0]
  g0[3] = (unsigned)((ga >> 32) & 0x1FFFFFFu) | (2u << 30); // ga[56:32] | type=2
  i32x8 g1;
  g1[0] = (int)(1u << 16);          // data_size=1 -> 2 bytes
  g1[1] = (int)(rowlen << 16);      // tensor_dim0[15:0]  @bit48
  g1[2] = (int)(rows << 16);        // tensor_dim1[15:0]  @bit80
  g1[3] = (int)(rowlen << 16);      // tile_dim0          @bit112
  g1[4] = (int)rows;                // tile_dim1          @bit128
  g1[5] = (int)stride;              // tensor_dim0_stride @bit160
  g1[6] = 0;
  g1[7] = 0;
  i32x4 z4 = {0, 0, 0, 0};
  i32x8 z8 = {0, 0, 0, 0, 0, 0, 0, 0};
  __builtin_amdgcn_tensor_load_to_lds(g0, g1, z4, z4, z8, 0);
}

// =========================================================================
// Pre-pass 0a: fp32 -> bf16 bulk convert (vectorized)
// =========================================================================
__global__ __launch_bounds__(256)
void cvt_f32_bf16_kernel(const float* __restrict__ src, bf16* __restrict__ dst) {
  size_t i = ((size_t)blockIdx.x * 256 + threadIdx.x) * 4;
  v4f v = *(const v4f*)(src + i);
  v4bf o;
#pragma unroll
  for (int e = 0; e < 4; ++e) o[e] = (bf16)v[e];
  *(v4bf*)(dst + i) = o;
}

// =========================================================================
// Pre-pass 0b: LDS-tiled transpose fp32[R][C] -> bf16[C][R]
// =========================================================================
__global__ __launch_bounds__(256)
void transpose_f32_bf16_kernel(const float* __restrict__ src, bf16* __restrict__ dst,
                               int R, int C) {
  __shared__ float tile[32][33];
  const int tx = threadIdx.x & 31, ty = threadIdx.x >> 5;   // 32x8
  const int nTc = C >> 5;
  const int r0 = (blockIdx.x / nTc) * 32;
  const int c0 = (blockIdx.x % nTc) * 32;
#pragma unroll
  for (int i = 0; i < 32; i += 8)
    tile[ty + i][tx] = src[(size_t)(r0 + ty + i) * C + c0 + tx];
  __syncthreads();
#pragma unroll
  for (int i = 0; i < 32; i += 8)
    dst[(size_t)(c0 + ty + i) * R + r0 + tx] = (bf16)tile[tx][ty + i];
}

// =========================================================================
// Kernel 1: qkv = x @ W_qkv^T (+RoPE), ping-pong double-buffered K-loop
// (unrolled x2 so buffer hand-off needs no register copies).
// =========================================================================
__global__ __launch_bounds__(256)
void qkv_rope_kernel(const bf16* __restrict__ xb, const bf16* __restrict__ Wtq,
                     bf16* __restrict__ qb, bf16* __restrict__ ktb,
                     bf16* __restrict__ vb) {
  const int lane = threadIdx.x & 31;
  const int wid  = threadIdx.x >> 5;
  const int wg   = blockIdx.x * 8 + wid;
  const int t0   = (wg / 48) * 16;
  const int e0   = (wg % 48) * 64;
  const int BS   = 3 * DM;

  v8f acc[4];
#pragma unroll
  for (int nb = 0; nb < 4; ++nb) acc[nb] = zero8();

  const bf16* apx = xb + t0 * DM;
  auto ldA = [&](int d0) { return load_a16(apx + d0, DM, lane); };
  auto ldB = [&](int d0, v16bf* bb) {
    const bf16* bp = Wtq + (size_t)d0 * BS + e0;
#pragma unroll
    for (int nb = 0; nb < 4; ++nb) bb[nb] = load_b16(bp + nb * 16, BS, lane);
  };
  auto domm = [&](v16bf a, v16bf* bb) {
#pragma unroll
    for (int nb = 0; nb < 4; ++nb) acc[nb] = wmma_bf16(a, bb[nb], acc[nb]);
  };

  v16bf a0 = ldA(0), a1;
  v16bf b0[4], b1[4];
  ldB(0, b0);
  int d0 = 0;
  for (; d0 < DM - 64; d0 += 64) {
    __builtin_prefetch(Wtq + (size_t)(d0 + 128) * BS + e0, 0, 3);
    a1 = ldA(d0 + 32); ldB(d0 + 32, b1);
    domm(a0, b0);
    a0 = ldA(d0 + 64); ldB(d0 + 64, b0);
    domm(a1, b1);
  }
  a1 = ldA(DM - 32); ldB(DM - 32, b1);
  domm(a0, b0);
  domm(a1, b1);

  const int region = e0 >> 10;            // 0=q 1=k 2=v
  const int h      = (e0 & 1023) >> 6;
  const int n      = lane & 15;
  const int rbase  = (lane >> 4) << 3;
  const float ln10000 = 9.210340371976184f;

#pragma unroll
  for (int nb = 0; nb < 4; ++nb) {
    const int j = nb * 16 + n;            // 0..63 within head
    float ov[8];
#pragma unroll
    for (int v = 0; v < 8; ++v) {
      const int t = t0 + rbase + v;
      float val = acc[nb][v];
      if (region < 2) {
        const int jm = j & 31;
        float inv_freq = __expf(-ln10000 * (float)(2 * jm) * (1.0f / 64.0f));
        float s, c;
        __sincosf((float)t * inv_freq, &s, &c);
        float pair = acc[nb ^ 2][v];      // rotate-half partner, same lane
        ov[v] = (j < 32) ? (c * val - s * pair) : (c * val + s * pair);
      } else {
        ov[v] = val;
      }
    }
    if (region == 1) {
      v8bf pk;
#pragma unroll
      for (int v = 0; v < 8; ++v) pk[v] = (bf16)ov[v];
      *(v8bf*)(ktb + ((size_t)h * DH + j) * T_SEQ + t0 + rbase) = pk;
    } else {
      bf16* dst = (region == 0) ? qb : vb;
#pragma unroll
      for (int v = 0; v < 8; ++v)
        dst[(size_t)h * T_SEQ * DH + (size_t)(t0 + rbase + v) * DH + j] = (bf16)ov[v];
    }
  }
}

// =========================================================================
// Kernel 2: causal flash attention. V tiles via TDM -> LDS (overlapped),
// K^T fragments ping-pong double-buffered (no rotation copies).
// =========================================================================
__global__ __launch_bounds__(256)
void flash_attn_kernel(const bf16* __restrict__ qb, const bf16* __restrict__ ktb,
                       const bf16* __restrict__ vb, bf16* __restrict__ ob) {
  __shared__ __align__(16) float sS[8][16][33];
  __shared__ __align__(16) bf16  sP[8][16][32];
  __shared__ float sRed[8][16];
  __shared__ __align__(128) bf16 sV[8][32][DH];   // 4KB per wave

  const int lane = threadIdx.x & 31;
  const int wid  = threadIdx.x >> 5;
  const int wg   = blockIdx.x * 8 + wid;
  const int h    = wg >> 8;
  const int q0   = (wg & 255) * 16;

  const bf16* Q  = qb  + (size_t)h * T_SEQ * DH;
  const bf16* KT = ktb + (size_t)h * DH * T_SEQ;  // [dh][T]
  const bf16* V  = vb  + (size_t)h * T_SEQ * DH;
  const unsigned sV_off = (unsigned)(uintptr_t)(&sV[wid][0][0]);

  const v16bf aq0 = load_a16(Q + q0 * DH,      DH, lane);
  const v16bf aq1 = load_a16(Q + q0 * DH + 32, DH, lane);

  v8f accO[4];
#pragma unroll
  for (int nb = 0; nb < 4; ++nb) accO[nb] = zero8();

  float m_i = -3.0e38f, l_i = 0.f;   // lanes 0..15 own row = lane
  const int n     = lane & 15;
  const int rbase = (lane >> 4) << 3;

  const int kblocks = (q0 + 16 + 31) >> 5;
  const int k0max   = (kblocks - 1) * 32;

  // bk[half*2+c]: B frag of K^T, keys k0+half*16.., dh chunk c*32..
  auto load_bk = [&](int k0, v16bf* bk) {
    bk[0] = load_b16(KT + k0,                            T_SEQ, lane);
    bk[1] = load_b16(KT + (size_t)32 * T_SEQ + k0,       T_SEQ, lane);
    bk[2] = load_b16(KT + k0 + 16,                       T_SEQ, lane);
    bk[3] = load_b16(KT + (size_t)32 * T_SEQ + k0 + 16,  T_SEQ, lane);
  };

  // one key-block: consumes bkc (K^T frags for k0), prefetches bkn for k0n
  auto step = [&](int k0, v16bf* bkc, int k0n, v16bf* bkn) {
    // previous block's sV ds reads must be done before TDM overwrites
    wave_lds_fence();
    tdm_load_2d_bf16(sV_off, V + (size_t)k0 * DH, 32, DH, DH);

    // ---- S = (Q.K^T) * 1/sqrt(dh) ----
#pragma unroll
    for (int half = 0; half < 2; ++half) {
      v8f s = zero8();
      s = wmma_bf16(aq0, bkc[half * 2 + 0], s);
      s = wmma_bf16(aq1, bkc[half * 2 + 1], s);
#pragma unroll
      for (int v = 0; v < 8; ++v)
        sS[wid][rbase + v][half * 16 + n] = s[v] * 0.125f;
    }
    // prefetch next block's K^T frags; overlaps softmax + PV below
    load_bk(k0n, bkn);

    wave_lds_fence();

    // ---- online softmax: lane r<16 owns query row q0+r ----
    if (lane < 16) {
      const int r = lane;
      int nv = q0 + r - k0 + 1;
      nv = nv < 0 ? 0 : (nv > 32 ? 32 : nv);
      float alpha = 1.f;
      if (nv > 0) {
        float mx = m_i;
        for (int c = 0; c < nv; ++c) mx = fmaxf(mx, sS[wid][r][c]);
        alpha = __expf(m_i - mx);
        float sum = 0.f;
        for (int c = 0; c < 32; ++c) {
          float p = (c < nv) ? __expf(sS[wid][r][c] - mx) : 0.f;
          sP[wid][r][c] = (bf16)p;
          sum += p;
        }
        l_i = l_i * alpha + sum;
        m_i = mx;
      } else {
        for (int c = 0; c < 32; ++c) sP[wid][r][c] = (bf16)0.f;
      }
      sRed[wid][r] = alpha;
    }
    wave_lds_fence();

    // ---- rescale O, then O += P.V (V from LDS, filled by TDM) ----
#pragma unroll
    for (int v = 0; v < 8; ++v) {
      float a = sRed[wid][rbase + v];
#pragma unroll
      for (int nb = 0; nb < 4; ++nb) accO[nb][v] *= a;
    }
    v16bf ap = load_a16(&sP[wid][0][0], 32, lane);

    __builtin_amdgcn_s_wait_tensorcnt(0);
    asm volatile("" ::: "memory");
#pragma unroll
    for (int nb = 0; nb < 4; ++nb) {
      v16bf bv = load_b16(&sV[wid][0][nb * 16], DH, lane);
      accO[nb] = wmma_bf16(ap, bv, accO[nb]);
    }
  };

  v16bf bkA[4], bkB[4];
  load_bk(0, bkA);
  int kbi = 0;
  for (; kbi + 1 < kblocks; kbi += 2) {
    const int k1 = (kbi + 1) * 32;
    int k2 = (kbi + 2) * 32;
    if (k2 > k0max) k2 = k0max;
    step(kbi * 32, bkA, k1, bkB);
    step(k1, bkB, k2, bkA);
  }
  if (kbi < kblocks)
    step(kbi * 32, bkA, k0max, bkB);

  if (lane < 16) sRed[wid][lane] = 1.f / l_i;
  wave_lds_fence();

#pragma unroll
  for (int v = 0; v < 8; ++v) {
    const float linv = sRed[wid][rbase + v];
    const int t = q0 + rbase + v;
#pragma unroll
    for (int nb = 0; nb < 4; ++nb)
      ob[(size_t)t * DM + h * DH + nb * 16 + n] = (bf16)(accO[nb][v] * linv);
  }
}

// =========================================================================
// Kernel 3: out = attn @ W_out^T, ping-pong double-buffered K-loop.
// =========================================================================
__global__ __launch_bounds__(256)
void out_proj_kernel(const bf16* __restrict__ ab, const bf16* __restrict__ Wto,
                     float* __restrict__ out) {
  const int lane = threadIdx.x & 31;
  const int wid  = threadIdx.x >> 5;
  const int wg   = blockIdx.x * 8 + wid;
  const int t0   = (wg >> 4) * 16;
  const int e0   = (wg & 15) * 64;

  v8f acc[4];
#pragma unroll
  for (int nb = 0; nb < 4; ++nb) acc[nb] = zero8();

  const bf16* apb = ab + t0 * DM;
  auto ldA = [&](int d0) { return load_a16(apb + d0, DM, lane); };
  auto ldB = [&](int d0, v16bf* bb) {
    const bf16* bp = Wto + (size_t)d0 * DM + e0;
#pragma unroll
    for (int nb = 0; nb < 4; ++nb) bb[nb] = load_b16(bp + nb * 16, DM, lane);
  };
  auto domm = [&](v16bf a, v16bf* bb) {
#pragma unroll
    for (int nb = 0; nb < 4; ++nb) acc[nb] = wmma_bf16(a, bb[nb], acc[nb]);
  };

  v16bf a0 = ldA(0), a1;
  v16bf b0[4], b1[4];
  ldB(0, b0);
  int d0 = 0;
  for (; d0 < DM - 64; d0 += 64) {
    __builtin_prefetch(Wto + (size_t)(d0 + 128) * DM + e0, 0, 3);
    a1 = ldA(d0 + 32); ldB(d0 + 32, b1);
    domm(a0, b0);
    a0 = ldA(d0 + 64); ldB(d0 + 64, b0);
    domm(a1, b1);
  }
  a1 = ldA(DM - 32); ldB(DM - 32, b1);
  domm(a0, b0);
  domm(a1, b1);

  const int n     = lane & 15;
  const int rbase = (lane >> 4) << 3;
#pragma unroll
  for (int v = 0; v < 8; ++v) {
    const int t = t0 + rbase + v;
#pragma unroll
    for (int nb = 0; nb < 4; ++nb)
      out[(size_t)t * DM + e0 + nb * 16 + n] = acc[nb][v];
  }
}

// =========================================================================
extern "C" void kernel_launch(void* const* d_in, const int* in_sizes, int n_in,
                              void* d_out, int out_size, void* d_ws, size_t ws_size,
                              hipStream_t stream) {
  (void)in_sizes; (void)n_in; (void)out_size; (void)ws_size;
  const float* x    = (const float*)d_in[0];   // [1,4096,1024]
  const float* Wqkv = (const float*)d_in[1];   // [3072,1024]
  const float* Wout = (const float*)d_in[2];   // [1024,1024]
  float* out = (float*)d_out;

  bf16* p = (bf16*)d_ws;
  bf16* xb  = p; p += (size_t)T_SEQ * DM;            // 4.0M elems
  bf16* Wtq = p; p += (size_t)DM * 3 * DM;           // 3.0M
  bf16* Wto = p; p += (size_t)DM * DM;               // 1.0M
  bf16* qb  = p; p += (size_t)NHEAD * T_SEQ * DH;    // 4.0M
  bf16* ktb = p; p += (size_t)NHEAD * DH * T_SEQ;    // 4.0M
  bf16* vb  = p; p += (size_t)NHEAD * T_SEQ * DH;    // 4.0M
  bf16* ab  = p;                                     // 4.0M  (~48MB total)

  cvt_f32_bf16_kernel<<<(T_SEQ * DM) / 1024, 256, 0, stream>>>(x, xb);
  transpose_f32_bf16_kernel<<<(3 * DM / 32) * (DM / 32), 256, 0, stream>>>(
      Wqkv, Wtq, 3 * DM, DM);
  transpose_f32_bf16_kernel<<<(DM / 32) * (DM / 32), 256, 0, stream>>>(
      Wout, Wto, DM, DM);

  qkv_rope_kernel<<<1536, 256, 0, stream>>>(xb, Wtq, qb, ktb, vb);
  flash_attn_kernel<<<512, 256, 0, stream>>>(qb, ktb, vb, ab);
  out_proj_kernel<<<512, 256, 0, stream>>>(ab, Wto, out);
}
